// GRAND_79413945303607
// MI455X (gfx1250) — matrix operand order repaired and checked by reading, not compiled, not run on hardware
//
#include <hip/hip_runtime.h>
#include <hip/hip_bf16.h>

// ---------------------------------------------------------------------------
// GRAND forward for MI455X (gfx1250, wave32).
//   1) x0 = X*0.5 ; prop = x0
//   2) 8 rounds: x1 = scatter_add(x0[src]*w, dst) via f32 L2 atomics; prop += x1
//   3) Xp = rownorm(prop/9) -> bf16
//   4) H  = relu(Xp @ W0 + b0)   [v_wmma_f32_16x16x32_bf16], accumulate row sumsq
//   5) out = (H @ W1) * (1/||H||_row) + b1   [v_wmma_f32_16x16x32_bf16]
//      (row-normalization folded into the f32 accumulator: matmul is linear)
// ---------------------------------------------------------------------------

typedef __bf16 bf16_t;
typedef __attribute__((ext_vector_type(16))) __bf16 v16bf;
typedef __attribute__((ext_vector_type(8)))  float  v8f;

#define NF   64          // input features
#define NH   256         // hidden
#define NC   47          // classes
#define NCP  48          // padded classes (3 x 16-wide N tiles)

__device__ __forceinline__ unsigned short f2bf(float f) {
  union { float f; unsigned u; } x; x.f = f;
  unsigned r = x.u + 0x7FFFu + ((x.u >> 16) & 1u);   // round-to-nearest-even
  return (unsigned short)(r >> 16);
}

// 16 bf16 elements viewed as vector / shorts / two 16-byte chunks
union Frag16 {
  v16bf v;
  unsigned short s[16];
  uint4 q[2];
};

// ---------------------------------------------------------------------------
// 1) init: x0 = X * (1 - dropnode) ; prop = x0   (vectorized float4)
__global__ void k_init(const float* __restrict__ X, float* __restrict__ x0,
                       float* __restrict__ prop, long long n4) {
  long long i = blockIdx.x * (long long)blockDim.x + threadIdx.x;
  if (i >= n4) return;
  float4 v = ((const float4*)X)[i];
  v.x *= 0.5f; v.y *= 0.5f; v.z *= 0.5f; v.w *= 0.5f;
  ((float4*)x0)[i] = v;
  ((float4*)prop)[i] = v;
}

// 2) scatter: one thread per (edge, 4-feature chunk); gather float4, 4 atomics.
//    16 lanes cover one 256B feature row -> fully coalesced gather; the x
//    buffers (25.6 MB) live in the 192 MB L2, so the f32 atomics stay on-chip.
__global__ void k_scatter(const float* __restrict__ x,
                          const int* __restrict__ src,
                          const int* __restrict__ dst,
                          const float* __restrict__ w,
                          float* __restrict__ y, int E) {
  long long tid = blockIdx.x * (long long)blockDim.x + threadIdx.x;
  int e = (int)(tid >> 4);
  int c = ((int)tid & 15) << 2;
  if (e >= E) return;
  int s = src[e];
  int d = dst[e];
  float we = w[e];
  float4 v = *(const float4*)(x + (long long)s * NF + c);
  float* o = y + (long long)d * NF + c;
  atomicAdd(o + 0, v.x * we);
  atomicAdd(o + 1, v.y * we);
  atomicAdd(o + 2, v.z * we);
  atomicAdd(o + 3, v.w * we);
}

// prop += x1
__global__ void k_accum(float* __restrict__ prop, const float* __restrict__ x,
                        long long n4) {
  long long i = blockIdx.x * (long long)blockDim.x + threadIdx.x;
  if (i >= n4) return;
  float4 a = ((const float4*)prop)[i];
  float4 b = ((const float4*)x)[i];
  a.x += b.x; a.y += b.y; a.z += b.z; a.w += b.w;
  ((float4*)prop)[i] = a;
}

// 3) row L2-normalize prop/9 over 64 features -> bf16. One wave32 per row,
//    each lane owns 2 consecutive floats; full-wave shfl_xor reduction.
__global__ void k_norm_xp(const float* __restrict__ prop,
                          unsigned short* __restrict__ Xp, int n) {
  int wave = (blockIdx.x * blockDim.x + threadIdx.x) >> 5;
  int lane = threadIdx.x & 31;
  if (wave >= n) return;
  const float inv = 1.0f / 9.0f;
  float2 v = *(const float2*)(prop + (long long)wave * NF + lane * 2);
  v.x *= inv; v.y *= inv;
  float s = v.x * v.x + v.y * v.y;
  #pragma unroll
  for (int off = 16; off; off >>= 1) s += __shfl_xor(s, off, 32);
  float sc = 1.0f / (1e-12f + sqrtf(s));
  unsigned short* o = Xp + (long long)wave * NF + lane * 2;
  o[0] = f2bf(v.x * sc);
  o[1] = f2bf(v.y * sc);
}

// transpose W0[64x256] -> col-major bf16 W0t[col][k], k contiguous
__global__ void k_convW0(const float* __restrict__ W0, unsigned short* __restrict__ W0t) {
  int t = blockIdx.x * blockDim.x + threadIdx.x;   // 64*256
  if (t >= NF * NH) return;
  int k = t / NH, col = t % NH;
  W0t[col * NF + k] = f2bf(W0[t]);
}

// transpose W1[256x47] -> col-major bf16 W1t[col][k], zero-pad cols >= 47
__global__ void k_convW1(const float* __restrict__ W1, unsigned short* __restrict__ W1t) {
  int t = blockIdx.x * blockDim.x + threadIdx.x;   // 48*256
  if (t >= NCP * NH) return;
  int col = t / NH, k = t % NH;
  float v = (col < NC) ? W1[k * NC + col] : 0.0f;
  W1t[col * NH + k] = f2bf(v);
}

// ---------------------------------------------------------------------------
// 4) GEMM1: H = relu(Xp @ W0 + b0).  M=n, K=64, N=256.
//    One wave per 16x16 tile; block = 8 waves covering 8 N-tiles;
//    grid = (n/16, 2). Also accumulates per-row sum of squares into rownorm.
//
//    A (16x32 bf16, lane l -> row l&15, K-block l>>4): per lane two contiguous
//    8-half groups at k0+kb*8 and k0+16+kb*8 (16B loads).
//    B (32x16 bf16, lane l -> col l&15): same k groups as A, k0+kb*8 and k0+16+kb*8 (col-major W).
//    C/D f32: c[r] -> row r + 8*(l>>4), col l&15.
__global__ void k_gemm1(const unsigned short* __restrict__ Xp,
                        const unsigned short* __restrict__ W0t,
                        const float* __restrict__ b0,
                        unsigned short* __restrict__ H,
                        float* __restrict__ rownorm, int n) {
  int lane = threadIdx.x & 31;
  int wv   = threadIdx.x >> 5;
  int mt   = blockIdx.x;
  int nt   = blockIdx.y * 8 + wv;
  int m16  = lane & 15;
  int kb   = lane >> 4;
  int row0 = mt * 16;
  if (row0 >= n) return;
  int col  = nt * 16 + m16;

  v8f c = {};
  const unsigned short* ap = Xp  + (long long)(row0 + m16) * NF;
  const unsigned short* bp = W0t + (long long)col * NF;
  #pragma unroll
  for (int kt = 0; kt < 2; ++kt) {
    int k0 = kt * 32;
    Frag16 a, b;
    a.q[0] = *(const uint4*)(ap + k0 + kb * 8);
    a.q[1] = *(const uint4*)(ap + k0 + 16 + kb * 8);
    b.q[0] = *(const uint4*)(bp + k0 + kb * 8);
    b.q[1] = *(const uint4*)(bp + k0 + 16 + kb * 8);
    c = __builtin_amdgcn_wmma_f32_16x16x32_bf16(false, a.v, false, b.v,
                                                (short)0, c, false, false);
  }

  float bias = b0[col];
  #pragma unroll
  for (int r = 0; r < 8; ++r) {
    int rl = r + 8 * kb;
    float v = c[r] + bias;
    v = v > 0.0f ? v : 0.0f;                       // relu
    H[(long long)(row0 + rl) * NH + col] = f2bf(v);
    float s = v * v;                               // partial row sumsq over 16 cols
    #pragma unroll
    for (int off = 8; off; off >>= 1) s += __shfl_xor(s, off, 16);
    if (m16 == 0) atomicAdd(rownorm + row0 + rl, s);
  }
}

// ---------------------------------------------------------------------------
// 5) GEMM2: out = (H @ W1) * inv_norm_row + b1.  M=n, K=256, N=47 (3 tiles).
//    Block = 3 waves (one per N-tile), grid = n/16. Raw bf16 H feeds the WMMA;
//    the per-row 1/||h|| is applied to the f32 accumulator. Each lane computes
//    the scale for row m16; D-rows r+8*kb fetch theirs via width-16 shuffles
//    (lane rl holds row rl's scale in both half-waves). Shuffles run before
//    the col<47 mask so bpermute sources stay active.
__global__ void k_gemm2(const unsigned short* __restrict__ H,
                        const float* __restrict__ rownorm,
                        const unsigned short* __restrict__ W1t,
                        const float* __restrict__ b1,
                        float* __restrict__ out, int n) {
  int lane = threadIdx.x & 31;
  int nt   = threadIdx.x >> 5;          // 0..2
  int mt   = blockIdx.x;
  int m16  = lane & 15;
  int kb   = lane >> 4;
  int row0 = mt * 16;
  if (row0 >= n) return;
  int col  = nt * 16 + m16;

  // this lane's row scale (row index m16 of the tile)
  float scm = 1.0f / (1e-12f + sqrtf(rownorm[row0 + m16]));

  v8f c = {};
  const unsigned short* ap = H   + (long long)(row0 + m16) * NH;
  const unsigned short* bp = W1t + (long long)col * NH;
  #pragma unroll
  for (int kt = 0; kt < 8; ++kt) {
    int k0 = kt * 32;
    Frag16 a, b;
    a.q[0] = *(const uint4*)(ap + k0 + kb * 8);
    a.q[1] = *(const uint4*)(ap + k0 + 16 + kb * 8);
    b.q[0] = *(const uint4*)(bp + k0 + kb * 8);
    b.q[1] = *(const uint4*)(bp + k0 + 16 + kb * 8);
    c = __builtin_amdgcn_wmma_f32_16x16x32_bf16(false, a.v, false, b.v,
                                                (short)0, c, false, false);
  }

  // broadcast row scales to D-fragment rows (unmasked: all lanes active)
  float scr[8];
  #pragma unroll
  for (int r = 0; r < 8; ++r) scr[r] = __shfl(scm, r + 8 * kb, 16);

  if (col < NC) {
    float bias = b1[col];
    #pragma unroll
    for (int r = 0; r < 8; ++r) {
      int rl = r + 8 * kb;
      out[(long long)(row0 + rl) * NC + col] = c[r] * scr[r] + bias;
    }
  }
}

// ---------------------------------------------------------------------------
extern "C" void kernel_launch(void* const* d_in, const int* in_sizes, int n_in,
                              void* d_out, int out_size, void* d_ws, size_t ws_size,
                              hipStream_t stream) {
  const float* X   = (const float*)d_in[0];
  const int*   src = (const int*)  d_in[1];
  const int*   dst = (const int*)  d_in[2];
  const float* ew  = (const float*)d_in[3];
  const float* W0  = (const float*)d_in[4];
  const float* b0  = (const float*)d_in[5];
  const float* W1  = (const float*)d_in[6];
  const float* b1  = (const float*)d_in[7];
  float* out = (float*)d_out;

  const int n = in_sizes[0] / NF;       // 100000 nodes
  const int E = in_sizes[1];            // 1.6M edges
  const long long nf  = (long long)n * NF;   // node-feature floats
  const long long nf4 = nf / 4;

  // --- workspace carve-up (256B aligned) ---
  auto al = [](size_t x) { return (x + 255) & ~(size_t)255; };
  char* base = (char*)d_ws;
  size_t off = 0;
  float* x0   = (float*)(base + off); off = al(off + nf * 4);
  float* x1   = (float*)(base + off); off = al(off + nf * 4);
  float* prop = (float*)(base + off); off = al(off + nf * 4);
  unsigned short* Xp  = (unsigned short*)(base + off); off = al(off + nf * 2);
  unsigned short* Hb  = (unsigned short*)(base + off); off = al(off + (size_t)n * NH * 2);
  float* rownorm      = (float*)(base + off);          off = al(off + (size_t)n * 4);
  unsigned short* W0t = (unsigned short*)(base + off); off = al(off + (size_t)NF * NH * 2);
  unsigned short* W1t = (unsigned short*)(base + off); off = al(off + (size_t)NCP * NH * 2);
  (void)ws_size; (void)n_in; (void)out_size;

  // --- weight conversion (independent of propagation) ---
  k_convW0<<<(NF * NH + 255) / 256, 256, 0, stream>>>(W0, W0t);
  k_convW1<<<(NCP * NH + 255) / 256, 256, 0, stream>>>(W1, W1t);

  // --- init + 8 propagation rounds ---
  k_init<<<(int)((nf4 + 255) / 256), 256, 0, stream>>>(X, x0, prop, nf4);

  float* xa = x0;
  float* xb = x1;
  const long long sthreads = (long long)E * 16;
  const int sblocks = (int)((sthreads + 255) / 256);
  for (int r = 0; r < 8; ++r) {
    hipMemsetAsync(xb, 0, nf * 4, stream);
    k_scatter<<<sblocks, 256, 0, stream>>>(xa, src, dst, ew, xb, E);
    k_accum<<<(int)((nf4 + 255) / 256), 256, 0, stream>>>(prop, xb, nf4);
    float* t = xa; xa = xb; xb = t;
  }

  // --- normalize -> bf16 features ---
  k_norm_xp<<<(n * 32 + 255) / 256, 256, 0, stream>>>(prop, Xp, n);

  // --- dense head on WMMA ---
  hipMemsetAsync(rownorm, 0, (size_t)n * 4, stream);
  dim3 g1((n + 15) / 16, 2);
  k_gemm1<<<g1, 256, 0, stream>>>(Xp, W0t, b0, Hb, rownorm, n);
  k_gemm2<<<(n + 15) / 16, 96, 0, stream>>>(Hb, rownorm, W1t, b1, out, n);
}
